// HybridFusionNetworkWithUncertainty_27530740367911
// MI455X (gfx1250) — matrix operand order repaired and checked by reading, not compiled, not run on hardware
//
#include <hip/hip_runtime.h>
#include <hip/hip_bf16.h>
#include <math.h>

// MI455X / gfx1250, wave32. Matrix math via v_wmma_f32_16x16x32_bf16 (fp32
// accumulate). Compute-bound (~93 GFLOP vs ~300MB HBM traffic), so bf16 WMMA
// is the right precision; fp32 for LayerNorm/softmax/sigmoid.
//
// Pipeline per K-iteration (single LDS buffer, register double-buffering):
//   s_wait_asynccnt 0 ; barrier            -> LDS(k) visible
//   ds_load ALL fragments (A + 4/8 B) -> regs (one dscnt wait total)
//   barrier                                -> all reads done, LDS reusable
//   global_load_async_to_lds  B(k+1)       -> overlaps with WMMA chain
//   v_wmma x4/8 back-to-back on registers

#define USE_ASYNC_LDS 1

typedef __attribute__((ext_vector_type(16))) __bf16 v16bf;
typedef __attribute__((ext_vector_type(8)))  float  v8f;

union BF16Frag { v16bf v; unsigned u[8]; };

#define KC   32
#define KPAD 40     // LDS row stride in bf16 elems (80B rows: 16B-aligned, bank-spread)
#define UWf  0.1f
#define EPSf 1e-5f

// ---- CDNA5 async global->LDS (ISA 15.18.3 op 96-98, ASYNCcnt tracked) -----
__device__ __forceinline__ void async_ld_b128(void* lds, const void* g) {
  asm volatile("global_load_async_to_lds_b128 %0, %1, off"
               :: "v"((unsigned)(unsigned long long)lds),
                  "v"((unsigned long long)g)
               : "memory");
}
__device__ __forceinline__ void async_ld_b64(void* lds, const void* g) {
  asm volatile("global_load_async_to_lds_b64 %0, %1, off"
               :: "v"((unsigned)(unsigned long long)lds),
                  "v"((unsigned long long)g)
               : "memory");
}
__device__ __forceinline__ void async_wait0() {
  asm volatile("s_wait_asynccnt 0x0" ::: "memory");
}

// ---- fragment loaders (ISA 7.12.2 layouts, wave32) ------------------------
__device__ __forceinline__ v16bf load_a_frag(const __bf16* As, int mrow, int lane) {
  BF16Frag a;
  const int m   = lane & 15;
  const int klo = (lane >> 4) * 8;
  const __bf16* ar = As + (mrow + m) * KPAD;
#pragma unroll
  for (int x = 0; x < 4; ++x) a.u[x]     = *(const unsigned*)(ar + klo + 2 * x);
#pragma unroll
  for (int x = 0; x < 4; ++x) a.u[4 + x] = *(const unsigned*)(ar + klo + 16 + 2 * x);
  return a.v;
}
__device__ __forceinline__ v16bf load_b_frag(const __bf16* Bs, int nrow, int lane) {
  BF16Frag b;
  const int n  = lane & 15;
  const int kb = (lane >> 4) * 8;
  const __bf16* br = Bs + (nrow + n) * KPAD;
#pragma unroll
  for (int x = 0; x < 8; ++x) b.u[x] = *(const unsigned*)(br + kb + 2 * x + ((x >> 2) << 3));
  return b.v;
}

// ---- weight pre-conversion ------------------------------------------------
__global__ __launch_bounds__(256) void cvt_f32_bf16_kernel(const float* __restrict__ src,
                                                           __bf16* __restrict__ dst, int n) {
  const int i = blockIdx.x * 256 + threadIdx.x;
  if (i < n) dst[i] = (__bf16)src[i];
}
__global__ __launch_bounds__(256) void cvt_pad_w1_kernel(const float* __restrict__ src,
                                                         __bf16* __restrict__ dst) {
  const int i = blockIdx.x * 256 + threadIdx.x;
  if (i < 256 * 1056) {
    const int r = i / 1056, c = i - r * 1056;
    const float v = (c < 1028) ? src[r * 1028 + c] : 0.f;
    dst[i] = (__bf16)v;
  }
}

// ---- Kernel A: H = LayerNorm(ReLU(X @ W^T + b)), bf16 out -----------------
// 256 thr / 8 waves, 32 rows x 512 cols per block; wave: rowtile (w&1),
// 128-col group (w>>1), 8 acc tiles. H row layout = b*4 + model.
__global__ __launch_bounds__(256) void intra_gemm_ln_kernel(
    const float* __restrict__ X, const __bf16* __restrict__ Wb,
    const float* __restrict__ bias, const float* __restrict__ lng,
    const float* __restrict__ lnb, __bf16* __restrict__ Hout,
    int d, int model) {
  __shared__ __bf16 As[32 * KPAD];
  __shared__ __bf16 Bs[512 * KPAD];
  __shared__ float  sums[32][4];
  __shared__ float  sqs[32][4];

  const int tid = threadIdx.x, w = tid >> 5, lane = tid & 31;
  const int rt = (w & 1) * 16;
  const int cg = w >> 1;
  const int colbase = cg * 128;
  const int row0 = blockIdx.x * 32;
  // staging coords: A = 4 elems/thread of 32x32; B = 2 rows/thread of 512x32
  const int sar = (tid * 4) >> 5, sac = (tid * 4) & 31;
  const int bn0 = tid * 2;

  const v8f z8 = {0.f, 0.f, 0.f, 0.f, 0.f, 0.f, 0.f, 0.f};
  v8f acc[8];
#pragma unroll
  for (int t = 0; t < 8; ++t) acc[t] = z8;

  // prologue: stage k0 = 0
  {
    const float4 f = *(const float4*)(X + (size_t)(row0 + sar) * d + sac);
    __bf16* dst = &As[sar * KPAD + sac];
    dst[0] = (__bf16)f.x; dst[1] = (__bf16)f.y; dst[2] = (__bf16)f.z; dst[3] = (__bf16)f.w;
#pragma unroll
    for (int rr = 0; rr < 2; ++rr) {
      const int n = bn0 + rr;
#if USE_ASYNC_LDS
#pragma unroll
      for (int j = 0; j < 4; ++j)
        async_ld_b128(&Bs[n * KPAD + 8 * j], Wb + (size_t)n * d + 8 * j);
#else
      const uint4* src = (const uint4*)(Wb + (size_t)n * d);
      uint4* bd = (uint4*)&Bs[n * KPAD];
      bd[0] = src[0]; bd[1] = src[1]; bd[2] = src[2]; bd[3] = src[3];
#endif
    }
  }

  for (int k0 = 0; k0 < d; k0 += KC) {
#if USE_ASYNC_LDS
    async_wait0();
#endif
    __syncthreads();                             // LDS(k0) ready for all waves
    const v16bf a = load_a_frag(As, rt, lane);
    v16bf bfr[8];
#pragma unroll
    for (int t = 0; t < 8; ++t) bfr[t] = load_b_frag(Bs, colbase + t * 16, lane);
    __syncthreads();                             // all frag reads done -> LDS reusable

    const bool more = (k0 + KC) < d;
    float4 af;
#if !USE_ASYNC_LDS
    uint4 breg[2][4];
#endif
    if (more) {
      const int k1 = k0 + KC;
#pragma unroll
      for (int rr = 0; rr < 2; ++rr) {
        const int n = bn0 + rr;
#if USE_ASYNC_LDS
#pragma unroll
        for (int j = 0; j < 4; ++j)
          async_ld_b128(&Bs[n * KPAD + 8 * j], Wb + (size_t)n * d + k1 + 8 * j);
#else
        const uint4* src = (const uint4*)(Wb + (size_t)n * d + k1);
#pragma unroll
        for (int j = 0; j < 4; ++j) breg[rr][j] = src[j];
#endif
      }
      af = *(const float4*)(X + (size_t)(row0 + sar) * d + k1 + sac);
    }
#pragma unroll
    for (int t = 0; t < 8; ++t)                  // back-to-back WMMA chain
      acc[t] = __builtin_amdgcn_wmma_f32_16x16x32_bf16(false, a, false, bfr[t],
                                                       (short)0, acc[t], false, false);
    if (more) {                                  // cvt+store after the chain
      __bf16* dst = &As[sar * KPAD + sac];
      dst[0] = (__bf16)af.x; dst[1] = (__bf16)af.y; dst[2] = (__bf16)af.z; dst[3] = (__bf16)af.w;
#if !USE_ASYNC_LDS
#pragma unroll
      for (int rr = 0; rr < 2; ++rr) {
        uint4* bd = (uint4*)&Bs[(bn0 + rr) * KPAD];
#pragma unroll
        for (int j = 0; j < 4; ++j) bd[j] = breg[rr][j];
      }
#endif
    }
  }

  // epilogue: bias + relu, LayerNorm over 512 cols (cross-wave via shfl+LDS)
  const int hi = lane >> 4, nl = lane & 15;
  float sm[8], sq[8];
#pragma unroll
  for (int r = 0; r < 8; ++r) { sm[r] = 0.f; sq[r] = 0.f; }
#pragma unroll
  for (int t = 0; t < 8; ++t) {
    const int col = colbase + t * 16 + nl;
    const float bc = bias[col];
#pragma unroll
    for (int r = 0; r < 8; ++r) {
      float v = acc[t][r] + bc;
      v = v > 0.f ? v : 0.f;
      acc[t][r] = v;
      sm[r] += v; sq[r] += v * v;
    }
  }
#pragma unroll
  for (int r = 0; r < 8; ++r) {
    float a1 = sm[r], a2 = sq[r];
#pragma unroll
    for (int m = 1; m < 16; m <<= 1) { a1 += __shfl_xor(a1, m, 32); a2 += __shfl_xor(a2, m, 32); }
    if (nl == 0) { const int row = rt + hi * 8 + r; sums[row][cg] = a1; sqs[row][cg] = a2; }
  }
  __syncthreads();
#pragma unroll
  for (int r = 0; r < 8; ++r) {
    const int row = rt + hi * 8 + r;
    const float tot  = sums[row][0] + sums[row][1] + sums[row][2] + sums[row][3];
    const float tot2 = sqs[row][0] + sqs[row][1] + sqs[row][2] + sqs[row][3];
    const float mean = tot * (1.f / 512.f);
    const float var  = tot2 * (1.f / 512.f) - mean * mean;
    const float rstd = rsqrtf(var + EPSf);
    __bf16* dst = Hout + ((size_t)(row0 + row) * 4 + model) * 512;
#pragma unroll
    for (int t = 0; t < 8; ++t) {
      const int col = colbase + t * 16 + nl;
      const float v = (acc[t][r] - mean) * rstd * lng[col] + lnb[col];
      dst[col] = (__bf16)v;
    }
  }
}

// ---- Kernel B: Q/V projections + 4x4 uncertainty attention, fused ---------
__global__ __launch_bounds__(256) void qv_attn_kernel(
    const __bf16* __restrict__ H, const __bf16* __restrict__ Wqb,
    const __bf16* __restrict__ Wvb, const float* __restrict__ bq,
    const float* __restrict__ bv, const float* __restrict__ pvars,
    const float* __restrict__ pscores, __bf16* __restrict__ Z) {
  __shared__ __bf16 As[32 * KPAD];
  __shared__ __bf16 Bs[256 * KPAD];
  __shared__ __bf16 Qs[32 * 264];
  __shared__ __bf16 Vs[32 * 264];

  const int tid = threadIdx.x, w = tid >> 5, lane = tid & 31;
  const int rt = (w & 1) * 16;
  const int colbase = (w >> 1) * 64;
  const int row0 = blockIdx.x * 32;
  const int hi = lane >> 4, nl = lane & 15;
  const int sar = (tid * 4) >> 5, sac = (tid * 4) & 31;
  const v8f z8 = {0.f, 0.f, 0.f, 0.f, 0.f, 0.f, 0.f, 0.f};

#pragma unroll
  for (int sel = 0; sel < 2; ++sel) {
    const __bf16* Wb = sel ? Wvb : Wqb;
    const float*  bb = sel ? bv : bq;
    v8f acc[4];
#pragma unroll
    for (int t = 0; t < 4; ++t) acc[t] = z8;

    // prologue: stage k0 = 0 (A and B both bf16 -> pure async path)
#if USE_ASYNC_LDS
    async_ld_b64(&As[sar * KPAD + sac], H + (size_t)(row0 + sar) * 512 + sac);
#pragma unroll
    for (int j = 0; j < 4; ++j)
      async_ld_b128(&Bs[tid * KPAD + 8 * j], Wb + (size_t)tid * 512 + 8 * j);
#else
    *(uint2*)&As[sar * KPAD + sac] = *(const uint2*)(H + (size_t)(row0 + sar) * 512 + sac);
    {
      const uint4* src = (const uint4*)(Wb + (size_t)tid * 512);
      uint4* bd = (uint4*)&Bs[tid * KPAD];
      bd[0] = src[0]; bd[1] = src[1]; bd[2] = src[2]; bd[3] = src[3];
    }
#endif

    for (int k0 = 0; k0 < 512; k0 += KC) {
#if USE_ASYNC_LDS
      async_wait0();
#endif
      __syncthreads();
      const v16bf a = load_a_frag(As, rt, lane);
      v16bf bfr[4];
#pragma unroll
      for (int t = 0; t < 4; ++t) bfr[t] = load_b_frag(Bs, colbase + t * 16, lane);
      __syncthreads();

      const bool more = (k0 + KC) < 512;
#if !USE_ASYNC_LDS
      uint2 areg; uint4 breg[4];
#endif
      if (more) {
        const int k1 = k0 + KC;
#if USE_ASYNC_LDS
        async_ld_b64(&As[sar * KPAD + sac], H + (size_t)(row0 + sar) * 512 + k1 + sac);
#pragma unroll
        for (int j = 0; j < 4; ++j)
          async_ld_b128(&Bs[tid * KPAD + 8 * j], Wb + (size_t)tid * 512 + k1 + 8 * j);
#else
        areg = *(const uint2*)(H + (size_t)(row0 + sar) * 512 + k1 + sac);
        const uint4* src = (const uint4*)(Wb + (size_t)tid * 512 + k1);
#pragma unroll
        for (int j = 0; j < 4; ++j) breg[j] = src[j];
#endif
      }
#pragma unroll
      for (int t = 0; t < 4; ++t)
        acc[t] = __builtin_amdgcn_wmma_f32_16x16x32_bf16(false, a, false, bfr[t],
                                                         (short)0, acc[t], false, false);
#if !USE_ASYNC_LDS
      if (more) {
        *(uint2*)&As[sar * KPAD + sac] = areg;
        uint4* bd = (uint4*)&Bs[tid * KPAD];
#pragma unroll
        for (int j = 0; j < 4; ++j) bd[j] = breg[j];
      }
#endif
    }
    __bf16* Dst = sel ? Vs : Qs;
#pragma unroll
    for (int t = 0; t < 4; ++t) {
      const int col = colbase + t * 16 + nl;
      const float bc = bb[col];
#pragma unroll
      for (int r = 0; r < 8; ++r)
        Dst[(rt + hi * 8 + r) * 264 + col] = (__bf16)(acc[t][r] + bc);
    }
  }
  __syncthreads();

  // attention: wave w owns batch item b (reference bug: key == query)
  {
    const size_t b = (size_t)blockIdx.x * 8 + w;
    float qr[4][8], vr[4][8];
#pragma unroll
    for (int i = 0; i < 4; ++i)
#pragma unroll
      for (int c8 = 0; c8 < 8; ++c8) {
        const int c = lane + 32 * c8;
        qr[i][c8] = (float)Qs[(w * 4 + i) * 264 + c];
        vr[i][c8] = (float)Vs[(w * 4 + i) * 264 + c];
      }
    const float* pv = pvars + b * 4;
    const float pvr[4] = {pv[0], pv[1], pv[2], pv[3]};
    float sc[4][4];
#pragma unroll
    for (int i = 0; i < 4; ++i)
#pragma unroll
      for (int j = 0; j < 4; ++j) {
        float p = 0.f;
#pragma unroll
        for (int c8 = 0; c8 < 8; ++c8) p += qr[i][c8] * qr[j][c8];
#pragma unroll
        for (int m = 1; m < 32; m <<= 1) p += __shfl_xor(p, m, 32);
        sc[i][j] = p - UWf * (pvr[i] + pvr[j]);
      }
    __bf16* zrow = Z + b * 1056;
#pragma unroll
    for (int i = 0; i < 4; ++i) {
      const float mx = fmaxf(fmaxf(sc[i][0], sc[i][1]), fmaxf(sc[i][2], sc[i][3]));
      float e0 = expf(sc[i][0] - mx), e1 = expf(sc[i][1] - mx);
      float e2 = expf(sc[i][2] - mx), e3 = expf(sc[i][3] - mx);
      const float rs = 1.f / (e0 + e1 + e2 + e3);
      e0 *= rs; e1 *= rs; e2 *= rs; e3 *= rs;
#pragma unroll
      for (int c8 = 0; c8 < 8; ++c8) {
        const float zv = e0 * vr[0][c8] + e1 * vr[1][c8] + e2 * vr[2][c8] + e3 * vr[3][c8];
        zrow[i * 256 + lane + 32 * c8] = (__bf16)zv;
      }
    }
    {  // cols 1024..1055: pred_scores then zero pad (matches zero-padded W1)
      const float t = (lane < 4) ? pscores[b * 4 + lane] : 0.f;
      zrow[1024 + lane] = (__bf16)t;
    }
  }
}

// ---- Kernel C: out1 = Z @ W1^T + b1 ; logit = out1 . Wfc + bfc ; sigmoid --
__global__ __launch_bounds__(256) void final_kernel(
    const __bf16* __restrict__ Z, const __bf16* __restrict__ W1b,
    const float* __restrict__ b1, const float* __restrict__ Wfc,
    const float* __restrict__ bfc, float* __restrict__ out) {
  __shared__ __bf16 As[32 * KPAD];
  __shared__ __bf16 Bs[256 * KPAD];
  __shared__ float  part[32][4];

  const int tid = threadIdx.x, w = tid >> 5, lane = tid & 31;
  const int rt = (w & 1) * 16;
  const int cg = w >> 1;
  const int colbase = cg * 64;
  const int row0 = blockIdx.x * 32;
  const int hi = lane >> 4, nl = lane & 15;
  const int sar = (tid * 4) >> 5, sac = (tid * 4) & 31;

  const v8f z8 = {0.f, 0.f, 0.f, 0.f, 0.f, 0.f, 0.f, 0.f};
  v8f acc[4];
#pragma unroll
  for (int t = 0; t < 4; ++t) acc[t] = z8;

#if USE_ASYNC_LDS
  async_ld_b64(&As[sar * KPAD + sac], Z + (size_t)(row0 + sar) * 1056 + sac);
#pragma unroll
  for (int j = 0; j < 4; ++j)
    async_ld_b128(&Bs[tid * KPAD + 8 * j], W1b + (size_t)tid * 1056 + 8 * j);
#else
  *(uint2*)&As[sar * KPAD + sac] = *(const uint2*)(Z + (size_t)(row0 + sar) * 1056 + sac);
  {
    const uint4* src = (const uint4*)(W1b + (size_t)tid * 1056);
    uint4* bd = (uint4*)&Bs[tid * KPAD];
    bd[0] = src[0]; bd[1] = src[1]; bd[2] = src[2]; bd[3] = src[3];
  }
#endif

  for (int k0 = 0; k0 < 1056; k0 += KC) {
#if USE_ASYNC_LDS
    async_wait0();
#endif
    __syncthreads();
    const v16bf a = load_a_frag(As, rt, lane);
    v16bf bfr[4];
#pragma unroll
    for (int t = 0; t < 4; ++t) bfr[t] = load_b_frag(Bs, colbase + t * 16, lane);
    __syncthreads();

    const bool more = (k0 + KC) < 1056;
#if !USE_ASYNC_LDS
    uint2 areg; uint4 breg[4];
#endif
    if (more) {
      const int k1 = k0 + KC;
#if USE_ASYNC_LDS
      async_ld_b64(&As[sar * KPAD + sac], Z + (size_t)(row0 + sar) * 1056 + k1 + sac);
#pragma unroll
      for (int j = 0; j < 4; ++j)
        async_ld_b128(&Bs[tid * KPAD + 8 * j], W1b + (size_t)tid * 1056 + k1 + 8 * j);
#else
      areg = *(const uint2*)(Z + (size_t)(row0 + sar) * 1056 + k1 + sac);
      const uint4* src = (const uint4*)(W1b + (size_t)tid * 1056 + k1);
#pragma unroll
      for (int j = 0; j < 4; ++j) breg[j] = src[j];
#endif
    }
#pragma unroll
    for (int t = 0; t < 4; ++t)
      acc[t] = __builtin_amdgcn_wmma_f32_16x16x32_bf16(false, a, false, bfr[t],
                                                       (short)0, acc[t], false, false);
#if !USE_ASYNC_LDS
    if (more) {
      *(uint2*)&As[sar * KPAD + sac] = areg;
      uint4* bd = (uint4*)&Bs[tid * KPAD];
#pragma unroll
      for (int j = 0; j < 4; ++j) bd[j] = breg[j];
    }
#endif
  }
  // fused GEMV: (acc + b1[col]) * Wfc[col] reduced to one logit per row
  float p[8];
#pragma unroll
  for (int r = 0; r < 8; ++r) p[r] = 0.f;
#pragma unroll
  for (int t = 0; t < 4; ++t) {
    const int col = colbase + t * 16 + nl;
    const float wf = Wfc[col], bb = b1[col];
#pragma unroll
    for (int r = 0; r < 8; ++r) p[r] += (acc[t][r] + bb) * wf;
  }
#pragma unroll
  for (int r = 0; r < 8; ++r) {
    float a1 = p[r];
#pragma unroll
    for (int m = 1; m < 16; m <<= 1) a1 += __shfl_xor(a1, m, 32);
    if (nl == 0) part[rt + hi * 8 + r][cg] = a1;
  }
  __syncthreads();
  if (tid < 32) {
    const float logit = part[tid][0] + part[tid][1] + part[tid][2] + part[tid][3] + bfc[0];
    out[row0 + tid] = 1.f / (1.f + expf(-logit));
  }
}

// ---------------------------------------------------------------------------
extern "C" void kernel_launch(void* const* d_in, const int* in_sizes, int n_in,
                              void* d_out, int out_size, void* d_ws, size_t ws_size,
                              hipStream_t stream) {
  (void)in_sizes; (void)n_in; (void)out_size; (void)ws_size;
  static const int FEAT[4] = {1024, 768, 512, 640};

  const float* lng     = (const float*)d_in[14];
  const float* lnb     = (const float*)d_in[15];
  const float* bq      = (const float*)d_in[17];
  const float* bv      = (const float*)d_in[21];
  const float* pscores = (const float*)d_in[12];
  const float* pvars   = (const float*)d_in[13];

  // workspace layout (segment offsets 16B aligned); total ~106 MB
  __bf16* Hbf = (__bf16*)d_ws;                          // 65536 x 512
  __bf16* Zbf = Hbf + (size_t)65536 * 512;              // 16384 x 1056 (padded)
  __bf16* Wib = Zbf + (size_t)16384 * 1056;             // 512 x 2944
  __bf16* Wqb = Wib + (size_t)512 * 2944;               // 256 x 512
  __bf16* Wvb = Wqb + (size_t)256 * 512;                // 256 x 512
  __bf16* W1b = Wvb + (size_t)256 * 512;                // 256 x 1056 (padded)

  // 0) pre-convert weights to bf16 (reused by hundreds of blocks each)
  size_t woff = 0;
  for (int i = 0; i < 4; ++i) {
    const int n = 512 * FEAT[i];
    cvt_f32_bf16_kernel<<<(n + 255) / 256, 256, 0, stream>>>(
        (const float*)d_in[1 + 3 * i], Wib + woff, n);
    woff += n;
  }
  cvt_f32_bf16_kernel<<<(256 * 512 + 255) / 256, 256, 0, stream>>>(
      (const float*)d_in[16], Wqb, 256 * 512);
  cvt_f32_bf16_kernel<<<(256 * 512 + 255) / 256, 256, 0, stream>>>(
      (const float*)d_in[20], Wvb, 256 * 512);
  cvt_pad_w1_kernel<<<(256 * 1056 + 255) / 256, 256, 0, stream>>>(
      (const float*)d_in[22], W1b);

  // 1) per-model GEMM + ReLU + LayerNorm -> Hbf (row = b*4 + model)
  woff = 0;
  for (int i = 0; i < 4; ++i) {
    intra_gemm_ln_kernel<<<16384 / 32, 256, 0, stream>>>(
        (const float*)d_in[3 * i], Wib + woff, (const float*)d_in[2 + 3 * i],
        lng, lnb, Hbf, FEAT[i], i);
    woff += (size_t)512 * FEAT[i];
  }

  // 2) fused Q/V projection + uncertainty attention -> Zbf (K padded to 1056)
  qv_attn_kernel<<<65536 / 32, 256, 0, stream>>>(
      Hbf, Wqb, Wvb, bq, bv, pvars, pscores, Zbf);

  // 3) final GEMM + GEMV + sigmoid -> d_out (16384 x 1 fp32)
  final_kernel<<<16384 / 32, 256, 0, stream>>>(
      Zbf, W1b, (const float*)d_in[23], (const float*)d_in[24],
      (const float*)d_in[25], (float*)d_out);
}
